// DeBERTa_78469052498053
// MI455X (gfx1250) — matrix operand order repaired and checked by reading, compiled, not run on hardware
//
#include <hip/hip_runtime.h>

typedef __attribute__((ext_vector_type(16))) __bf16 v16bf;
typedef __attribute__((ext_vector_type(8)))  __bf16 v8bf;
typedef __attribute__((ext_vector_type(8)))  float  v8f;

#define DB_B   8
#define DB_S   512
#define DB_H   768
#define DB_NH  12
#define DB_DH  64
#define DB_L   12
#define DB_FF  3072
#define DB_SPAN 256

#define BM 128
#define BN 64
#define BK 32
#define LDL 40   // padded LDS row stride (bf16 elems): 80B, 16B-aligned, bank-conflict free

union FragU { v16bf v; v8bf h[2]; };

// CDNA5 async global->LDS copy (ASYNCcnt-tracked, 16B per lane).
// vdst = 32-bit LDS byte address, vaddr = 64-bit global address.
__device__ __forceinline__ void async_b128(unsigned lds, const __bf16* g) {
  asm volatile("global_load_async_to_lds_b128 %0, %1, off"
               :: "v"(lds), "v"(g) : "memory");
}
__device__ __forceinline__ void wait_async0() {
  asm volatile("s_wait_asynccnt 0x0" ::: "memory");
}

// -------------------------------------------------------------------------
// Batched WMMA GEMM:  C[m,n] = sum_k A[m,k] * Bt[n,k]  (+ bias[n])
// A, Bt: bf16 row-major.  Per-z offsets: off = (z/bdiv)*s1 + (z%bdiv)*s2.
// MODE 0: fp32 out; 1: bf16 out; 2: per-head V-transpose bf16; 3: gelu->bf16
// Staging: double-buffered LDS filled by global_load_async_to_lds_b128.
// -------------------------------------------------------------------------
template<int MODE>
__global__ __launch_bounds__(256) void k_gemm(
    const __bf16* __restrict__ A, long long a1, long long a2, int lda,
    const __bf16* __restrict__ Bt, long long s1, long long s2, int ldb,
    float* __restrict__ C, __bf16* __restrict__ Cbf,
    long long c1, long long c2, int ldc,
    const float* __restrict__ bias, int K, int bdiv)
{
  __shared__ __bf16 As[2][BM * LDL];
  __shared__ __bf16 Bs[2][BN * LDL];

  const int z = blockIdx.z;
  const long long zq = z / bdiv, zr = z % bdiv;
  A  += zq * a1 + zr * a2;
  Bt += zq * s1 + zr * s2;
  const long long coff = zq * c1 + zr * c2;

  const int tid  = threadIdx.x;
  const int lane = tid & 31, wave = tid >> 5;
  const int wm = wave >> 1, wn = wave & 1;
  const int m0 = blockIdx.x * BM, n0 = blockIdx.y * BN;

  // staging: 256 threads, 16B (8 bf16) per thread per slot
  const int rA0 = tid >> 2;            // rows 0..63
  const int rA1 = rA0 + 64;            // rows 64..127 (A only)
  const int c8  = (tid & 3) << 3;      // k-offset 0/8/16/24
  const __bf16* Ag0 = A  + (long long)(m0 + rA0) * lda + c8;
  const __bf16* Ag1 = A  + (long long)(m0 + rA1) * lda + c8;
  const __bf16* Bg  = Bt + (long long)(n0 + rA0) * ldb + c8;

  unsigned lA0[2], lA1[2], lB[2];
  #pragma unroll
  for (int bu = 0; bu < 2; bu++) {
    lA0[bu] = (unsigned)(size_t)&As[bu][rA0 * LDL + c8];
    lA1[bu] = (unsigned)(size_t)&As[bu][rA1 * LDL + c8];
    lB [bu] = (unsigned)(size_t)&Bs[bu][rA0 * LDL + c8];
  }

  v8f acc[2][2];
  #pragma unroll
  for (int i = 0; i < 2; i++)
    #pragma unroll
    for (int j = 0; j < 2; j++)
      #pragma unroll
      for (int e = 0; e < 8; e++) acc[i][j][e] = 0.f;

  const int hl = lane >> 4, lr = lane & 15;
  const int nk = K / BK;

  // prologue: fill buffer 0
  async_b128(lA0[0], Ag0);
  async_b128(lA1[0], Ag1);
  async_b128(lB[0],  Bg);

  for (int kt = 0; kt < nk; ++kt) {
    const int cur = kt & 1;
    wait_async0();        // my async writes into buf[cur] have landed
    __syncthreads();      // everyone's have; everyone done reading buf[cur^1]
    if (kt + 1 < nk) {
      Ag0 += BK; Ag1 += BK; Bg += BK;
      async_b128(lA0[cur ^ 1], Ag0);
      async_b128(lA1[cur ^ 1], Ag1);
      async_b128(lB [cur ^ 1], Bg);
    }
    const __bf16* Ab = As[cur];
    const __bf16* Bb = Bs[cur];

    // A fragment: lane half hl reads K {hl*8..+7} and {hl*8+16..+23}
    FragU fa[2], fb[2];
    #pragma unroll
    for (int i = 0; i < 2; i++) {
      const __bf16* p = &Ab[(wm * 32 + i * 16 + lr) * LDL + hl * 8];
      fa[i].h[0] = *(const v8bf*)(p);
      fa[i].h[1] = *(const v8bf*)(p + 16);
    }
    // B fragment: lane half hl reads K {hl*8..+7} and {hl*8+16..+23}
    #pragma unroll
    for (int j = 0; j < 2; j++) {
      const __bf16* p = &Bb[(wn * 32 + j * 16 + lr) * LDL + hl * 8];
      fb[j].h[0] = *(const v8bf*)(p);
      fb[j].h[1] = *(const v8bf*)(p + 16);
    }
    #pragma unroll
    for (int i = 0; i < 2; i++)
      #pragma unroll
      for (int j = 0; j < 2; j++)
        acc[i][j] = __builtin_amdgcn_wmma_f32_16x16x32_bf16(
            false, fa[i].v, false, fb[j].v, (short)0, acc[i][j], false, false);
  }

  // epilogue: lane lr -> col, VGPR r -> row (r + hl*8)
  #pragma unroll
  for (int i = 0; i < 2; i++) {
    const int mb = m0 + wm * 32 + i * 16 + hl * 8;
    #pragma unroll
    for (int j = 0; j < 2; j++) {
      const int n = n0 + wn * 32 + j * 16 + lr;
      const float bv = bias ? bias[n] : 0.f;
      #pragma unroll
      for (int r = 0; r < 8; r++) {
        const int m = mb + r;
        float val = acc[i][j][r] + bv;
        if (MODE == 0) {
          C[coff + (long long)m * ldc + n] = val;
        } else if (MODE == 1) {
          Cbf[coff + (long long)m * ldc + n] = (__bf16)val;
        } else if (MODE == 2) {
          const int bb = m >> 9, ss = m & 511;   // m = b*S + s
          const int hh = n >> 6, dd = n & 63;    // n = h*DH + d
          Cbf[((long long)(bb * DB_NH + hh) * DB_DH + dd) * DB_S + ss] = (__bf16)val;
        } else {
          float gl = 0.5f * val * (1.f + erff(val * 0.70710678f));
          Cbf[coff + (long long)m * ldc + n] = (__bf16)gl;
        }
      }
    }
  }
}

// ----------------------------- small kernels -----------------------------

__global__ void k_convert(const float* __restrict__ s, __bf16* __restrict__ d, int n) {
  int i = blockIdx.x * 256 + threadIdx.x;
  if (i < n) d[i] = (__bf16)s[i];
}

// dst[n*K + k] = src[k*N + n]  (fp32 [K,N] -> bf16 [N,K])
__global__ void k_convert_wt(const float* __restrict__ s, __bf16* __restrict__ d, int K, int N) {
  int i = blockIdx.x * 256 + threadIdx.x;
  if (i >= K * N) return;
  int n = i / K, k = i - n * K;
  d[i] = (__bf16)s[(long long)k * N + n];
}

__global__ void k_embed(const float* __restrict__ we, const float* __restrict__ pe,
                        const int* __restrict__ ids, float* __restrict__ out) {
  long long i = (long long)blockIdx.x * 256 + threadIdx.x;  // < B*S*H
  int c = (int)(i % DB_H);
  long long r = i / DB_H;
  int s = (int)(r % DB_S);
  out[i] = we[(long long)ids[r] * DB_H + c] + pe[(long long)s * DB_H + c];
}

__device__ __forceinline__ int db_bucket(int i, int j) {
  int rel = i - j;
  const int mid = DB_SPAN / 2;  // 128
  bool sml = (rel < mid) && (rel > -mid);
  float ap = sml ? (float)(mid - 1) : fabsf((float)rel);
  float logp = ceilf(logf(ap / (float)mid) / logf(511.f / (float)mid) * (float)(mid - 1)) + (float)mid;
  int sgn = (rel > 0) - (rel < 0);
  return (ap <= (float)mid) ? rel : ((int)logp) * sgn;
}

__global__ void k_relpos(int* __restrict__ cidx, int* __restrict__ pidx) {
  int i = blockIdx.x * 256 + threadIdx.x;  // < S*S
  int q = i >> 9, k = i & 511;
  int c = db_bucket(q, k) + DB_SPAN;
  int p = -db_bucket(k, q) + DB_SPAN;
  cidx[i] = min(max(c, 0), 2 * DB_SPAN - 1);
  pidx[i] = min(max(p, 0), 2 * DB_SPAN - 1);
}

// LayerNorm(+optional residual, +optional row mask) -> fp32 and bf16
__global__ __launch_bounds__(256) void k_ln(
    const float* __restrict__ X, const float* __restrict__ R,
    const float* __restrict__ g, const float* __restrict__ b,
    const int* __restrict__ mask,
    float* __restrict__ outF, __bf16* __restrict__ outB)
{
  __shared__ float red[256];
  const long long row = blockIdx.x;
  const float* x = X + row * DB_H;
  const int t = threadIdx.x;
  float v0 = x[t], v1 = x[t + 256], v2 = x[t + 512];
  if (R) {
    const float* r = R + row * DB_H;
    v0 += r[t]; v1 += r[t + 256]; v2 += r[t + 512];
  }
  red[t] = v0 + v1 + v2; __syncthreads();
  for (int o = 128; o > 0; o >>= 1) { if (t < o) red[t] += red[t + o]; __syncthreads(); }
  float mu = red[0] * (1.f / DB_H);
  __syncthreads();
  float d0 = v0 - mu, d1 = v1 - mu, d2 = v2 - mu;
  red[t] = d0 * d0 + d1 * d1 + d2 * d2; __syncthreads();
  for (int o = 128; o > 0; o >>= 1) { if (t < o) red[t] += red[t + o]; __syncthreads(); }
  float inv = rsqrtf(red[0] * (1.f / DB_H) + 1e-7f);
  float mk = mask ? (float)mask[row] : 1.f;
  long long base = row * DB_H;
  float y0 = (d0 * inv * g[t] + b[t]) * mk;
  float y1 = (d1 * inv * g[t + 256] + b[t + 256]) * mk;
  float y2 = (d2 * inv * g[t + 512] + b[t + 512]) * mk;
  outF[base + t] = y0; outF[base + t + 256] = y1; outF[base + t + 512] = y2;
  outB[base + t] = (__bf16)y0; outB[base + t + 256] = (__bf16)y1; outB[base + t + 512] = (__bf16)y2;
}

// scores[z,q,k] = (scores + c2p[z,q,cidx[q,k]] + p2c[z,k,pidx[q,k]]) / sqrt(3*DH)
__global__ void k_mix(float* __restrict__ sc, const __bf16* __restrict__ c2p,
                      const __bf16* __restrict__ p2c,
                      const int* __restrict__ cidx, const int* __restrict__ pidx) {
  long long i = (long long)blockIdx.x * 256 + threadIdx.x;
  int qk = (int)(i & (long long)(DB_S * DB_S - 1));
  long long z = i >> 18;
  int q = qk >> 9, k = qk & 511;
  float v = sc[i]
          + (float)c2p[(z * DB_S + q) * (2 * DB_SPAN) + cidx[qk]]
          + (float)p2c[(z * DB_S + k) * (2 * DB_SPAN) + pidx[qk]];
  sc[i] = v * 0.07216878f;  // 1/sqrt(3*64)
}

// masked softmax over 512 cols, write bf16 probs (masked entries -> 0)
__global__ __launch_bounds__(256) void k_softmax(const float* __restrict__ sc,
    const int* __restrict__ am, __bf16* __restrict__ P) {
  __shared__ float red[256];
  const long long row = blockIdx.x;          // b*NH*S + h*S + q
  const int b = (int)(row / (DB_NH * DB_S));
  const int q = (int)(row % DB_S);
  const float* s = sc + row * DB_S;
  const int* mrow = am + (long long)b * DB_S;
  const int rv = mrow[q];
  const int t = threadIdx.x;
  float x0 = s[t], x1 = s[t + 256];
  bool g0 = rv && mrow[t];
  bool g1 = rv && mrow[t + 256];
  const float NEG = -3.0e38f;
  red[t] = fmaxf(g0 ? x0 : NEG, g1 ? x1 : NEG); __syncthreads();
  for (int o = 128; o > 0; o >>= 1) { if (t < o) red[t] = fmaxf(red[t], red[t + o]); __syncthreads(); }
  float mx = red[0]; __syncthreads();
  float e0 = g0 ? expf(x0 - mx) : 0.f;
  float e1 = g1 ? expf(x1 - mx) : 0.f;
  red[t] = e0 + e1; __syncthreads();
  for (int o = 128; o > 0; o >>= 1) { if (t < o) red[t] += red[t + o]; __syncthreads(); }
  float zs = red[0];
  float iz = zs > 0.f ? 1.f / zs : 0.f;
  P[row * DB_S + t] = (__bf16)(e0 * iz);
  P[row * DB_S + t + 256] = (__bf16)(e1 * iz);
}

// ----------------------------- host driver -----------------------------

extern "C" void kernel_launch(void* const* d_in, const int* in_sizes, int n_in,
                              void* d_out, int out_size, void* d_ws, size_t ws_size,
                              hipStream_t stream)
{
  const float* word_emb = (const float*)d_in[0];
  const float* pos_emb  = (const float*)d_in[1];
  const float* eg  = (const float*)d_in[2];
  const float* eb  = (const float*)d_in[3];
  const float* rel = (const float*)d_in[4];
  const float* Wq  = (const float*)d_in[5];
  const float* bq  = (const float*)d_in[6];
  const float* Wk  = (const float*)d_in[7];
  const float* bk  = (const float*)d_in[8];
  const float* Wv  = (const float*)d_in[9];
  const float* bv  = (const float*)d_in[10];
  const float* Wo  = (const float*)d_in[11];
  const float* bo  = (const float*)d_in[12];
  const float* g1  = (const float*)d_in[13];
  const float* b1p = (const float*)d_in[14];
  const float* Wi  = (const float*)d_in[15];
  const float* bi  = (const float*)d_in[16];
  const float* Wo2 = (const float*)d_in[17];
  const float* bo2 = (const float*)d_in[18];
  const float* g2  = (const float*)d_in[19];
  const float* b2p = (const float*)d_in[20];
  const int* ids = (const int*)d_in[21];
  const int* am  = (const int*)d_in[22];
  (void)in_sizes; (void)n_in; (void)out_size; (void)ws_size;

  const long long BS  = (long long)DB_B * DB_S;              // 4096
  const long long BSH = BS * DB_H;                           // 3145728
  const long long SS  = (long long)DB_S * DB_S;              // 262144
  const long long ZSS = (long long)DB_B * DB_NH * SS;        // 25165824
  const long long SH  = (long long)DB_S * DB_H;              // 393216

  char* w = (char*)d_ws; size_t off = 0;
  auto alloc = [&](size_t bytes) -> void* {
    void* p = w + off; off += (bytes + 255) & ~(size_t)255; return p;
  };

  float* hF  = (float*)d_out;                    // hidden state lives in d_out
  float* h1F = (float*)alloc(BSH * 4);
  float* tF  = (float*)alloc(BSH * 4);
  __bf16* hbf = (__bf16*)alloc(BSH * 2);
  __bf16* qbf = (__bf16*)alloc(BSH * 2);
  __bf16* kbf = (__bf16*)alloc(BSH * 2);
  __bf16* vT  = (__bf16*)alloc(BSH * 2);
  __bf16* relbf  = (__bf16*)alloc((size_t)512 * DB_H * 2);
  __bf16* poskbf = (__bf16*)alloc((size_t)512 * DB_H * 2);
  __bf16* posqbf = (__bf16*)alloc((size_t)512 * DB_H * 2);
  __bf16* Wqt  = (__bf16*)alloc((size_t)DB_H * DB_H * 2);
  __bf16* Wkt  = (__bf16*)alloc((size_t)DB_H * DB_H * 2);
  __bf16* Wvt  = (__bf16*)alloc((size_t)DB_H * DB_H * 2);
  __bf16* Wot  = (__bf16*)alloc((size_t)DB_H * DB_H * 2);
  __bf16* Wit  = (__bf16*)alloc((size_t)DB_H * DB_FF * 2);
  __bf16* Wo2t = (__bf16*)alloc((size_t)DB_H * DB_FF * 2);
  int* cidx = (int*)alloc(SS * 4);
  int* pidx = (int*)alloc(SS * 4);
  float*  scores = (float*)alloc(ZSS * 4);
  __bf16* c2pb   = (__bf16*)alloc(ZSS * 2);
  __bf16* p2cb   = (__bf16*)alloc(ZSS * 2);
  __bf16* probsb = p2cb;   // alias: softmax runs after k_mix consumed p2c
  __bf16* ffbf   = c2pb;   // alias: FFN runs after k_mix consumed c2p (25MB <= 50MB)
  __bf16* ctxbf  = qbf;    // alias: ctx written after q last read

  auto gemm = [&](int mode,
                  const __bf16* A, long long a1, long long a2, int lda,
                  const __bf16* Bt, long long s1, long long s2, int ldb,
                  float* C, __bf16* Cbf, long long c1, long long c2, int ldc,
                  const float* bias, int M, int N, int K, int Z, int bdiv) {
    dim3 grd(M / BM, N / BN, Z);
    if (mode == 0)      k_gemm<0><<<grd, 256, 0, stream>>>(A, a1, a2, lda, Bt, s1, s2, ldb, C, Cbf, c1, c2, ldc, bias, K, bdiv);
    else if (mode == 1) k_gemm<1><<<grd, 256, 0, stream>>>(A, a1, a2, lda, Bt, s1, s2, ldb, C, Cbf, c1, c2, ldc, bias, K, bdiv);
    else if (mode == 2) k_gemm<2><<<grd, 256, 0, stream>>>(A, a1, a2, lda, Bt, s1, s2, ldb, C, Cbf, c1, c2, ldc, bias, K, bdiv);
    else                k_gemm<3><<<grd, 256, 0, stream>>>(A, a1, a2, lda, Bt, s1, s2, ldb, C, Cbf, c1, c2, ldc, bias, K, bdiv);
  };

  // ---- prologue ----
  k_convert<<<(int)((512 * DB_H) / 256), 256, 0, stream>>>(rel, relbf, 512 * DB_H);
  k_relpos<<<(int)(SS / 256), 256, 0, stream>>>(cidx, pidx);
  k_embed<<<(int)(BSH / 256), 256, 0, stream>>>(word_emb, pos_emb, ids, tF);
  k_ln<<<(int)BS, 256, 0, stream>>>(tF, nullptr, eg, eb, am, hF, hbf);

  const long long HH = (long long)DB_H * DB_H;
  const long long HF = (long long)DB_H * DB_FF;

  for (int l = 0; l < DB_L; l++) {
    const float* Wq_l = Wq + l * HH;   const float* bq_l = bq + (long long)l * DB_H;
    const float* Wk_l = Wk + l * HH;   const float* bk_l = bk + (long long)l * DB_H;
    const float* Wv_l = Wv + l * HH;   const float* bv_l = bv + (long long)l * DB_H;
    const float* Wo_l = Wo + l * HH;   const float* bo_l = bo + (long long)l * DB_H;
    const float* Wi_l = Wi + l * HF;   const float* bi_l = bi + (long long)l * DB_FF;
    const float* Wo2_l = Wo2 + l * HF; const float* bo2_l = bo2 + (long long)l * DB_H;
    const float* g1_l = g1 + (long long)l * DB_H; const float* b1_l = b1p + (long long)l * DB_H;
    const float* g2_l = g2 + (long long)l * DB_H; const float* b2_l = b2p + (long long)l * DB_H;

    const int nW = DB_H * DB_H, nWi = DB_H * DB_FF;
    k_convert_wt<<<nW / 256, 256, 0, stream>>>(Wq_l, Wqt, DB_H, DB_H);
    k_convert_wt<<<nW / 256, 256, 0, stream>>>(Wk_l, Wkt, DB_H, DB_H);
    k_convert_wt<<<nW / 256, 256, 0, stream>>>(Wv_l, Wvt, DB_H, DB_H);
    k_convert_wt<<<nW / 256, 256, 0, stream>>>(Wo_l, Wot, DB_H, DB_H);
    k_convert_wt<<<nWi / 256, 256, 0, stream>>>(Wi_l, Wit, DB_H, DB_FF);
    k_convert_wt<<<nWi / 256, 256, 0, stream>>>(Wo2_l, Wo2t, DB_FF, DB_H);

    // Q, K, V projections (V written transposed per head: [B,NH,DH,S])
    gemm(1, hbf, 0, 0, DB_H, Wqt, 0, 0, DB_H, nullptr, qbf, 0, 0, DB_H, bq_l, (int)BS, DB_H, DB_H, 1, 1);
    gemm(1, hbf, 0, 0, DB_H, Wkt, 0, 0, DB_H, nullptr, kbf, 0, 0, DB_H, bk_l, (int)BS, DB_H, DB_H, 1, 1);
    gemm(2, hbf, 0, 0, DB_H, Wvt, 0, 0, DB_H, nullptr, vT,  0, 0, DB_H, bv_l, (int)BS, DB_H, DB_H, 1, 1);
    // shared-key/query position projections
    gemm(1, relbf, 0, 0, DB_H, Wkt, 0, 0, DB_H, nullptr, poskbf, 0, 0, DB_H, bk_l, 512, DB_H, DB_H, 1, 1);
    gemm(1, relbf, 0, 0, DB_H, Wqt, 0, 0, DB_H, nullptr, posqbf, 0, 0, DB_H, bq_l, 512, DB_H, DB_H, 1, 1);
    // scores = Q·K^T per (b,h)
    gemm(0, qbf, SH, 64, DB_H, kbf, SH, 64, DB_H,
         scores, nullptr, (long long)DB_NH * SS, SS, DB_S, nullptr,
         DB_S, DB_S, DB_DH, DB_B * DB_NH, DB_NH);
    // c2p = Q·pos_k^T,  p2c = K·pos_q^T (bf16)
    gemm(1, qbf, SH, 64, DB_H, poskbf, 0, 64, DB_H,
         nullptr, c2pb, (long long)DB_NH * SS, SS, 512, nullptr,
         DB_S, 512, DB_DH, DB_B * DB_NH, DB_NH);
    gemm(1, kbf, SH, 64, DB_H, posqbf, 0, 64, DB_H,
         nullptr, p2cb, (long long)DB_NH * SS, SS, 512, nullptr,
         DB_S, 512, DB_DH, DB_B * DB_NH, DB_NH);
    k_mix<<<(int)(ZSS / 256), 256, 0, stream>>>(scores, c2pb, p2cb, cidx, pidx);
    k_softmax<<<DB_B * DB_NH * DB_S, 256, 0, stream>>>(scores, am, probsb);
    // ctx = P·V  -> [B,S,H] bf16
    gemm(1, probsb, (long long)DB_NH * SS, SS, DB_S,
         vT, (long long)DB_NH * DB_DH * DB_S, (long long)DB_DH * DB_S, DB_S,
         nullptr, ctxbf, SH, 64, DB_H, nullptr,
         DB_S, DB_DH, DB_S, DB_B * DB_NH, DB_NH);
    // attention output projection -> pre-LN fp32
    gemm(0, ctxbf, 0, 0, DB_H, Wot, 0, 0, DB_H, tF, nullptr, 0, 0, DB_H, bo_l, (int)BS, DB_H, DB_H, 1, 1);
    k_ln<<<(int)BS, 256, 0, stream>>>(tF, hF, g1_l, b1_l, nullptr, h1F, hbf);
    // FFN
    gemm(3, hbf, 0, 0, DB_H, Wit, 0, 0, DB_H, nullptr, ffbf, 0, 0, DB_FF, bi_l, (int)BS, DB_FF, DB_H, 1, 1);
    gemm(0, ffbf, 0, 0, DB_FF, Wo2t, 0, 0, DB_FF, tF, nullptr, 0, 0, DB_H, bo2_l, (int)BS, DB_H, DB_FF, 1, 1);
    k_ln<<<(int)BS, 256, 0, stream>>>(tF, h1F, g2_l, b2_l, nullptr, hF, hbf);
  }
}
